// SelfAttention_21294447854159
// MI455X (gfx1250) — matrix operand order repaired and checked by reading, not compiled, not run on hardware
//
#include <hip/hip_runtime.h>

typedef unsigned short u16;
typedef unsigned int   u32;
typedef __attribute__((ext_vector_type(8)))  __bf16 v8bf;
typedef __attribute__((ext_vector_type(16))) __bf16 v16bf;
typedef __attribute__((ext_vector_type(8)))  float  v8f;

#define WMMA_BF16(a, b, c) \
  __builtin_amdgcn_wmma_f32_16x16x32_bf16(false, (a), false, (b), (short)0, (c), false, false)

#define D_HID  768
#define NHEAD  12
#define DHEAD  64
#define SEQ    2048
#define BATCH  4
#define ATT_SCALE 0.036084391824351615f  /* 768^-0.5 (reference scales by dim_hidden) */

static __device__ inline v8f v8f_zero() {
  v8f z;
  #pragma unroll
  for (int i = 0; i < 8; i++) z[i] = 0.0f;
  return z;
}

// ---- f32 -> (hi, lo) bf16 split: hi+lo carries ~16 mantissa bits -----------
static __device__ inline u16 bf16_rne(float f) {
  u32 u = __float_as_uint(f);
  return (u16)((u + 0x7FFFu + ((u >> 16) & 1u)) >> 16);
}
static __device__ inline float bf16_f32(u16 h) {
  return __uint_as_float(((u32)h) << 16);
}
static __device__ inline void split2(float f, u16& hi, u16& lo) {
  hi = bf16_rne(f);
  lo = bf16_rne(f - bf16_f32(hi));
}

// ---- fragment loads (contiguous k = 16*half + s, same for A and B) ----------------------------
static __device__ inline v16bf ld_bfrag(const u16* p) {        // 16 contiguous K
  v8bf a = *(const v8bf*)p;
  v8bf b = *(const v8bf*)(p + 8);
  return __builtin_shufflevector(a, b, 0,1,2,3,4,5,6,7,8,9,10,11,12,13,14,15);
}
static __device__ inline v16bf ld_afrag(const u16* row_kc, int half) {
  v8bf a = *(const v8bf*)(row_kc + half * 16);
  v8bf b = *(const v8bf*)(row_kc + half * 16 + 8);
  return __builtin_shufflevector(a, b, 0,1,2,3,4,5,6,7,8,9,10,11,12,13,14,15);
}

// ---- CDNA5 async global->LDS (ASYNCcnt) ------------------------------------
typedef __attribute__((address_space(3))) u16 as3_u16;
static __device__ inline u32 lds_addr(u16* p) {
  return (u32)(unsigned long long)(as3_u16*)p;   // LDS byte offset
}
static __device__ inline void async_b128(u32 lds, const void* gaddr) {
  asm volatile("global_load_async_to_lds_b128 %0, %1, off"
               :: "v"(lds), "v"(gaddr) : "memory");
}
#define WAIT_ASYNC(n) asm volatile("s_wait_asynccnt " #n ::: "memory")

// ---------------------------------------------------------------------------
// One-shot f32 -> split bf16 planes (row-major).
// ---------------------------------------------------------------------------
__global__ __launch_bounds__(256)
void split_plane(const float* __restrict__ src, u16* __restrict__ hi,
                 u16* __restrict__ lo, int n4) {
  int i = blockIdx.x * 256 + threadIdx.x;
  if (i >= n4) return;
  float4 v = ((const float4*)src)[i];
  u16 h0,h1,h2,h3,l0,l1,l2,l3;
  split2(v.x,h0,l0); split2(v.y,h1,l1); split2(v.z,h2,l2); split2(v.w,h3,l3);
  ((uint2*)hi)[i] = make_uint2((u32)h0 | ((u32)h1 << 16), (u32)h2 | ((u32)h3 << 16));
  ((uint2*)lo)[i] = make_uint2((u32)l0 | ((u32)l1 << 16), (u32)l2 | ((u32)l3 << 16));
}
// One-shot f32 [K,N] -> split bf16 planes transposed to [N,K].
__global__ __launch_bounds__(256)
void split_plane_T(const float* __restrict__ src, u16* __restrict__ hiT,
                   u16* __restrict__ loT, int K, int N) {
  int idx = blockIdx.x * 256 + threadIdx.x;
  if (idx >= K * N) return;
  int k = idx / N, n = idx % N;
  u16 h, l;
  split2(src[idx], h, l);
  hiT[(size_t)n * K + k] = h;
  loT[(size_t)n * K + k] = l;
}

// ---------------------------------------------------------------------------
// GEMM on pre-split bf16 planes: C = X @ W (+bias); W planes are K-major [N,K].
// Async double-buffered staging; 256 thr = 8 waves; tile 128(M) x 64(N);
// wave w: rows [64*(w&1)..+64), cols [16*(w>>1)..+16). K-chunk 32; fp32 accum.
// Output: fp32 Cf, or split planes Chi/Clo (row-major), with the V column
// range [1536,2304) optionally diverted to [b,h,dh,seq] planes VThi/VTlo.
// ---------------------------------------------------------------------------
__global__ __launch_bounds__(256)
void gemm_bf16sp(const u16* __restrict__ Xhi_g, const u16* __restrict__ Xlo_g,
                 const u16* __restrict__ WThi_g, const u16* __restrict__ WTlo_g,
                 const float* __restrict__ bias,
                 float* __restrict__ Cf,
                 u16* __restrict__ Chi, u16* __restrict__ Clo,
                 u16* __restrict__ VThi, u16* __restrict__ VTlo,
                 int M, int K, int N) {
  __shared__ u16 Xsh[2][128 * 40], Xsl[2][128 * 40];  // [m][k], stride 40
  __shared__ u16 Wsh[2][64 * 40],  Wsl[2][64 * 40];   // [n][k], stride 40

  const int t    = threadIdx.x;
  const int wave = t >> 5;
  const int lane = t & 31;
  const int half = lane >> 4;
  const int l16  = lane & 15;
  const int wm   = wave & 1;
  const int wn   = wave >> 1;
  const int m0   = blockIdx.y * 128;
  const int n0   = blockIdx.x * 64;

  auto issueTile = [&](int buf, int k0) {
    #pragma unroll
    for (int i = 0; i < 2; i++) {            // X: 512 chunks of 8 u16 / plane
      int flat8 = i * 256 + t;
      int row  = flat8 >> 2;
      int coff = (flat8 & 3) << 3;
      size_t g = (size_t)(m0 + row) * K + k0 + coff;
      async_b128(lds_addr(&Xsh[buf][row * 40 + coff]), Xhi_g + g);
      async_b128(lds_addr(&Xsl[buf][row * 40 + coff]), Xlo_g + g);
    }
    {                                        // W: 256 chunks / plane (K-major)
      int row  = t >> 2;
      int coff = (t & 3) << 3;
      size_t g = (size_t)(n0 + row) * K + k0 + coff;
      async_b128(lds_addr(&Wsh[buf][row * 40 + coff]), WThi_g + g);
      async_b128(lds_addr(&Wsl[buf][row * 40 + coff]), WTlo_g + g);
    }
  };

  v8f acc[4];
  #pragma unroll
  for (int i = 0; i < 4; i++) acc[i] = v8f_zero();

  const int nk = K / 32;
  issueTile(0, 0);
  for (int c = 0; c < nk; c++) {
    WAIT_ASYNC(0x0);          // tile c resident (this wave's chunks)
    __syncthreads();          // all waves' chunks visible; other buffer free
    if (c + 1 < nk) issueTile((c + 1) & 1, (c + 1) * 32);

    const u16* Xh = Xsh[c & 1];
    const u16* Xl = Xsl[c & 1];
    const u16* Wh = Wsh[c & 1];
    const u16* Wl = Wsl[c & 1];
    v16bf Bh = ld_bfrag(&Wh[(wn * 16 + l16) * 40 + half * 16]);
    v16bf Bl = ld_bfrag(&Wl[(wn * 16 + l16) * 40 + half * 16]);
    #pragma unroll
    for (int mt = 0; mt < 4; mt++) {
      const int row = wm * 64 + mt * 16 + l16;
      v16bf Ah = ld_afrag(&Xh[row * 40], half);
      v16bf Al = ld_afrag(&Xl[row * 40], half);
      acc[mt] = WMMA_BF16(Ah, Bl, acc[mt]);   // small terms first
      acc[mt] = WMMA_BF16(Al, Bh, acc[mt]);
      acc[mt] = WMMA_BF16(Ah, Bh, acc[mt]);
    }
  }

  // Epilogue. C layout: VGPR r -> row r (lanes 0-15) / r+8 (lanes 16-31).
  const int col = n0 + wn * 16 + l16;
  const float bv = bias ? bias[col] : 0.0f;
  const bool isV = (VThi != nullptr) && (n0 >= 2 * D_HID);
  #pragma unroll
  for (int mt = 0; mt < 4; mt++) {
    int row0 = m0 + wm * 64 + mt * 16 + half * 8;
    if (Cf) {
      #pragma unroll
      for (int r = 0; r < 8; r++)
        Cf[(size_t)(row0 + r) * N + col] = acc[mt][r] + bv;
    } else if (!isV) {
      #pragma unroll
      for (int r = 0; r < 8; r++) {
        u16 h, l;
        split2(acc[mt][r] + bv, h, l);
        size_t idx = (size_t)(row0 + r) * N + col;
        Chi[idx] = h;
        Clo[idx] = l;
      }
    } else {
      // V part -> [b, h, dh, seq] planes; 8 consecutive rows = consecutive n.
      int hh = (col - 2 * D_HID) >> 6;
      int d  = col & 63;
      int b_ = row0 >> 11;
      int nn = row0 & 2047;
      u16 hs[8], ls[8];
      #pragma unroll
      for (int r = 0; r < 8; r++) split2(acc[mt][r], hs[r], ls[r]);
      size_t base = ((size_t)(b_ * NHEAD + hh) * DHEAD + d) * SEQ + nn;
      *(uint4*)(VThi + base) = make_uint4(
          (u32)hs[0] | ((u32)hs[1] << 16), (u32)hs[2] | ((u32)hs[3] << 16),
          (u32)hs[4] | ((u32)hs[5] << 16), (u32)hs[6] | ((u32)hs[7] << 16));
      *(uint4*)(VTlo + base) = make_uint4(
          (u32)ls[0] | ((u32)ls[1] << 16), (u32)ls[2] | ((u32)ls[3] << 16),
          (u32)ls[4] | ((u32)ls[5] << 16), (u32)ls[6] | ((u32)ls[7] << 16));
    }
  }
}

// ---------------------------------------------------------------------------
// Flash attention on split bf16 planes. Async staggered staging: K issued
// after post-S barrier, V after post-PV barrier; in-order async completion
// makes s_wait_asynccnt 8 retire exactly the needed tile.
// One block (4 waves) per (b, h, 64-row q-block).
// ---------------------------------------------------------------------------
__global__ __launch_bounds__(128)
void flash_attn_bf16sp(const u16* __restrict__ qkvhi, const u16* __restrict__ qkvlo,
                       const u16* __restrict__ vthi, const u16* __restrict__ vtlo,
                       u16* __restrict__ yhi, u16* __restrict__ ylo) {
  __shared__ u16 Khs[64 * 72],  Kls[64 * 72];   // [key][dh]
  __shared__ u16 Vths[64 * 72], Vtls[64 * 72];  // [dh][key]
  __shared__ u16 Phs[64 * 72],  Pls[64 * 72];   // [qrow][key], wave-private rows

  const int t    = threadIdx.x;
  const int wave = t >> 5;
  const int lane = t & 31;
  const int half = lane >> 4;
  const int l16  = lane & 15;

  const int bh = blockIdx.x;
  const int b  = bh / NHEAD;
  const int h  = bh % NHEAD;
  const int q0 = blockIdx.y * 64;

  const size_t rs = 3 * D_HID;  // 2304
  const size_t base = (size_t)b * SEQ * rs + (size_t)h * DHEAD;
  const u16* Khg = qkvhi + base + D_HID;
  const u16* Klg = qkvlo + base + D_HID;
  const u16* VThg = vthi + (size_t)bh * DHEAD * SEQ;
  const u16* VTlg = vtlo + (size_t)bh * DHEAD * SEQ;

  auto issueK = [&](int kb) {
    #pragma unroll
    for (int i = 0; i < 4; i++) {
      int flat8 = i * 128 + t;
      int row  = flat8 >> 3;          // key
      int coff = (flat8 & 7) << 3;    // dh
      size_t g = (size_t)(kb + row) * rs + coff;
      async_b128(lds_addr(&Khs[row * 72 + coff]), Khg + g);
      async_b128(lds_addr(&Kls[row * 72 + coff]), Klg + g);
    }
  };
  auto issueV = [&](int kb) {
    #pragma unroll
    for (int i = 0; i < 4; i++) {
      int flat8 = i * 128 + t;
      int row  = flat8 >> 3;          // dh
      int coff = (flat8 & 7) << 3;    // key
      size_t g = (size_t)row * SEQ + kb + coff;
      async_b128(lds_addr(&Vths[row * 72 + coff]), VThg + g);
      async_b128(lds_addr(&Vtls[row * 72 + coff]), VTlg + g);
    }
  };

  // Preload this wave's Q A-fragments straight from the bf16 planes.
  v16bf qh[2], ql[2];
  {
    const u16* qhrow = qkvhi + base + (size_t)(q0 + wave * 16 + l16) * rs;
    const u16* qlrow = qkvlo + base + (size_t)(q0 + wave * 16 + l16) * rs;
    #pragma unroll
    for (int c = 0; c < 2; c++) {
      qh[c] = ld_afrag(qhrow + c * 32, half);
      ql[c] = ld_afrag(qlrow + c * 32, half);
    }
  }

  v8f o[4];
  #pragma unroll
  for (int i = 0; i < 4; i++) o[i] = v8f_zero();
  float m_r[8], l_r[8];
  #pragma unroll
  for (int r = 0; r < 8; r++) { m_r[r] = -3.0e38f; l_r[r] = 0.0f; }

  issueK(0);   // 8 async instrs
  issueV(0);   // 8 async instrs

  for (int kb0 = 0; kb0 < SEQ; kb0 += 64) {
    WAIT_ASYNC(0x8);   // K tile resident (V may still be in flight)
    __syncthreads();   // K visible to all waves

    // S = Q @ K^T : 16 rows x 64 keys; K-dim(dh)=64 = 2 chunks x 3 terms.
    v8f s[4];
    #pragma unroll
    for (int i = 0; i < 4; i++) s[i] = v8f_zero();
    #pragma unroll
    for (int c = 0; c < 2; c++) {
      int kc = c * 32;
      #pragma unroll
      for (int ct = 0; ct < 4; ct++) {
        v16bf Bh = ld_bfrag(&Khs[(ct * 16 + l16) * 72 + kc + half * 16]);
        v16bf Bl = ld_bfrag(&Kls[(ct * 16 + l16) * 72 + kc + half * 16]);
        s[ct] = WMMA_BF16(qh[c], Bl, s[ct]);
        s[ct] = WMMA_BF16(ql[c], Bh, s[ct]);
        s[ct] = WMMA_BF16(qh[c], Bh, s[ct]);
      }
    }
    __syncthreads();   // all waves done with K tile
    if (kb0 + 64 < SEQ) issueK(kb0 + 64);

    // Online softmax (element r is row half*8+r; xor 1/2/4/8 reduce the row).
    float corr[8];
    #pragma unroll
    for (int r = 0; r < 8; r++) {
      float mx = s[0][r] * ATT_SCALE;
      #pragma unroll
      for (int ct = 1; ct < 4; ct++) mx = fmaxf(mx, s[ct][r] * ATT_SCALE);
      #pragma unroll
      for (int msk = 1; msk < 16; msk <<= 1)
        mx = fmaxf(mx, __shfl_xor(mx, msk, 32));
      float mnew = fmaxf(m_r[r], mx);
      corr[r] = __expf(m_r[r] - mnew);
      m_r[r] = mnew;
      float sum = 0.0f;
      #pragma unroll
      for (int ct = 0; ct < 4; ct++) {
        float p = __expf(s[ct][r] * ATT_SCALE - mnew);
        s[ct][r] = p;
        sum += p;
      }
      #pragma unroll
      for (int msk = 1; msk < 16; msk <<= 1)
        sum += __shfl_xor(sum, msk, 32);
      l_r[r] = l_r[r] * corr[r] + sum;
    }
    #pragma unroll
    for (int nt = 0; nt < 4; nt++)
      #pragma unroll
      for (int r = 0; r < 8; r++) o[nt][r] *= corr[r];

    // Split P into this wave's private LDS rows (same-wave DS is in-order).
    #pragma unroll
    for (int ct = 0; ct < 4; ct++) {
      #pragma unroll
      for (int r = 0; r < 8; r++) {
        u16 ph, pl;
        split2(s[ct][r], ph, pl);
        int idx = (wave * 16 + half * 8 + r) * 72 + ct * 16 + l16;
        Phs[idx] = ph;
        Pls[idx] = pl;
      }
    }

    WAIT_ASYNC(0x8);   // V tile resident (K(i+1) still in flight)
    __syncthreads();   // V visible to all waves

    // O += P @ V : K-dim(keys)=64 = 2 chunks x 3 terms, 4 dh subtiles.
    #pragma unroll
    for (int c = 0; c < 2; c++) {
      int kc = c * 32;
      v16bf Ah = ld_afrag(&Phs[(wave * 16 + l16) * 72 + kc], half);
      v16bf Al = ld_afrag(&Pls[(wave * 16 + l16) * 72 + kc], half);
      #pragma unroll
      for (int nt = 0; nt < 4; nt++) {
        v16bf Bh = ld_bfrag(&Vths[(nt * 16 + l16) * 72 + kc + half * 16]);
        v16bf Bl = ld_bfrag(&Vtls[(nt * 16 + l16) * 72 + kc + half * 16]);
        o[nt] = WMMA_BF16(Ah, Bl, o[nt]);
        o[nt] = WMMA_BF16(Al, Bh, o[nt]);
        o[nt] = WMMA_BF16(Ah, Bh, o[nt]);
      }
    }
    __syncthreads();   // all waves done with V tile
    if (kb0 + 64 < SEQ) issueV(kb0 + 64);
  }

  // Epilogue: y = o / l, written as split bf16 planes.
  float inv_l[8];
  #pragma unroll
  for (int r = 0; r < 8; r++) inv_l[r] = 1.0f / l_r[r];
  #pragma unroll
  for (int nt = 0; nt < 4; nt++) {
    #pragma unroll
    for (int r = 0; r < 8; r++) {
      int n = q0 + wave * 16 + half * 8 + r;
      int d = nt * 16 + l16;
      u16 hh, ll;
      split2(o[nt][r] * inv_l[r], hh, ll);
      size_t idx = ((size_t)(b * SEQ + n)) * D_HID + h * DHEAD + d;
      yhi[idx] = hh;
      ylo[idx] = ll;
    }
  }
}

// ---------------------------------------------------------------------------
extern "C" void kernel_launch(void* const* d_in, const int* in_sizes, int n_in,
                              void* d_out, int out_size, void* d_ws, size_t ws_size,
                              hipStream_t stream) {
  const float* x     = (const float*)d_in[0];  // [4,2048,768]
  const float* W_in  = (const float*)d_in[1];  // [768,2304]
  const float* W_out = (const float*)d_in[2];  // [768,768]
  const float* b_out = (const float*)d_in[3];  // [768]
  float* out = (float*)d_out;                  // [4,2048,768]

  const int M = BATCH * SEQ;                   // 8192
  const size_t nx  = (size_t)M * D_HID;
  const size_t nwi = (size_t)D_HID * 3 * D_HID;
  const size_t nwo = (size_t)D_HID * D_HID;
  const size_t nq  = (size_t)M * 3 * D_HID;
  const size_t nv  = (size_t)BATCH * NHEAD * DHEAD * SEQ;
  const size_t ny  = (size_t)M * D_HID;

  u16* p    = (u16*)d_ws;
  u16* xhi  = p;            u16* xlo  = xhi + nx;
  u16* wihi = xlo + nx;     u16* wilo = wihi + nwi;   // [2304,768] K-major
  u16* wohi = wilo + nwi;   u16* wolo = wohi + nwo;   // [768,768]  K-major
  u16* qhi  = wolo + nwo;   u16* qlo  = qhi + nq;     // row-major (Q,K parts)
  u16* vthi = qlo + nq;     u16* vtlo = vthi + nv;    // [b,h,dh,seq]
  u16* yhi  = vtlo + nv;    u16* ylo  = yhi + ny;

  // 0) one-shot splits (W planes transposed to K-major)
  split_plane<<<dim3((u32)(nx / 4 + 255) / 256), dim3(256), 0, stream>>>(
      x, xhi, xlo, (int)(nx / 4));
  split_plane_T<<<dim3((u32)(nwi + 255) / 256), dim3(256), 0, stream>>>(
      W_in, wihi, wilo, D_HID, 3 * D_HID);
  split_plane_T<<<dim3((u32)(nwo + 255) / 256), dim3(256), 0, stream>>>(
      W_out, wohi, wolo, D_HID, D_HID);

  // 1) qkv = x @ W_in : Q,K -> row-major planes; V -> [b,h,dh,seq] planes
  gemm_bf16sp<<<dim3((3 * D_HID) / 64, M / 128), dim3(256), 0, stream>>>(
      xhi, xlo, wihi, wilo, nullptr, nullptr, qhi, qlo, vthi, vtlo,
      M, D_HID, 3 * D_HID);

  // 2) flash attention -> split bf16 y planes
  flash_attn_bf16sp<<<dim3(BATCH * NHEAD, SEQ / 64), dim3(128), 0, stream>>>(
      qhi, qlo, vthi, vtlo, yhi, ylo);

  // 3) out = y @ W_out + b_out (fp32)
  gemm_bf16sp<<<dim3(D_HID / 64, M / 128), dim3(256), 0, stream>>>(
      yhi, ylo, wohi, wolo, b_out, out, nullptr, nullptr, nullptr, nullptr,
      M, D_HID, D_HID);
}
